// Block_mamba_4999341932554
// MI455X (gfx1250) — matrix operand order repaired and checked by reading, not compiled, not run on hardware
//
#include <hip/hip_runtime.h>
#include <hip/hip_bf16.h>
#include <math.h>

// ---------------------------------------------------------------------------
// Problem constants (fixed by setup_inputs)
// ---------------------------------------------------------------------------
constexpr int kB   = 16;
constexpr int kH   = 56;
constexpr int kW   = 56;
constexpr int kN   = kH * kW;        // 3136 tokens per image
constexpr int kC   = 256;
constexpr int kCg  = 64;
constexpr int kL   = kN;             // scan length
constexpr int kHID = 1024;
constexpr long long kMT = (long long)kB * kN;   // 50176 total tokens

typedef __attribute__((ext_vector_type(16))) __bf16 v16bf;
typedef __attribute__((ext_vector_type(8)))  float  v8f;
typedef __attribute__((ext_vector_type(4)))  unsigned int u32x4;
typedef __attribute__((ext_vector_type(8)))  int i32x8;
typedef __attribute__((ext_vector_type(4)))  int i32x4;

// ---------------------------------------------------------------------------
// helpers
// ---------------------------------------------------------------------------
__device__ __forceinline__ unsigned short f2bf(float f) {
    union { float f; unsigned int u; } v; v.f = f;
    unsigned int r = v.u + 0x7FFFu + ((v.u >> 16) & 1u);   // round-to-nearest-even
    return (unsigned short)(r >> 16);
}
__device__ __forceinline__ float sigmoidf_(float x) { return 1.f / (1.f + expf(-x)); }
__device__ __forceinline__ float siluf_(float x)    { return x * sigmoidf_(x); }

// ---------------------------------------------------------------------------
// fp32 -> bf16 bulk convert (weights)
// ---------------------------------------------------------------------------
__global__ void cvt_bf16_kernel(const float* __restrict__ src,
                                unsigned short* __restrict__ dst, int n) {
    int i = blockIdx.x * 256 + threadIdx.x;
    if (i < n) dst[i] = f2bf(src[i]);
}

// ---------------------------------------------------------------------------
// TDM: stage a [rows x cols] bf16 tile (row stride = row_stride elements) into
// LDS at byte offset lds_off, with 16B padding per row (pitch = cols+8 elems)
// via D# pad_enable.  CDNA5 ISA.  2D tile -> groups 2/3 zero.
// ---------------------------------------------------------------------------
#if defined(__has_builtin)
#if __has_builtin(__builtin_amdgcn_tensor_load_to_lds) && __has_builtin(__builtin_amdgcn_s_wait_tensorcnt)
#define USE_TDM 1
#endif
#endif

#ifdef USE_TDM
__device__ __forceinline__ void tdm_load_tile(const unsigned short* gsrc,
                                              unsigned lds_off,
                                              int rows, int cols, int row_stride)
{
    unsigned long long ga = (unsigned long long)gsrc;
    u32x4 g0;
    g0[0] = 1u;                                           // count=1 (valid user D#)
    g0[1] = lds_off;                                      // lds_addr (bytes)
    g0[2] = (unsigned)(ga & 0xFFFFFFFFu);                 // global_addr[31:0]
    g0[3] = (unsigned)((ga >> 32) & 0x01FFFFFFu) | (2u << 30);  // addr[56:32] | type=2

    // pad: 16B every row -> interval code log2(rowBytes/8), amount code 3 (4 dw)
    const unsigned pic = (unsigned)(__ffs((cols * 2) >> 3) - 1);
    i32x8 g1;
    g1[0] = (int)((1u << 16) |                            // data_size = 2 bytes
                  (1u << 20) |                            // pad_enable
                  (pic << 22) | (3u << 25));              // pad_interval, pad_amount
    unsigned dim0 = (unsigned)row_stride;                 // tensor dim0 (elems)
    unsigned dim1 = (unsigned)rows;
    g1[1] = (int)((dim0 & 0xFFFFu) << 16);                // [63:48] = dim0 lo16
    g1[2] = (int)(((dim0 >> 16) & 0xFFFFu) | ((dim1 & 0xFFFFu) << 16));
    g1[3] = (int)(((dim1 >> 16) & 0xFFFFu) | (((unsigned)cols & 0xFFFFu) << 16)); // tile_dim0
    g1[4] = (int)(dim1 & 0xFFFFu);                        // tile_dim1 (tile_dim2=0)
    g1[5] = (int)(unsigned)row_stride;                    // tensor_dim0_stride[31:0]
    g1[6] = 0;
    g1[7] = 0;
    i32x4 gz = {0, 0, 0, 0};
#if __clang_major__ >= 23
    i32x8 gz8 = {0, 0, 0, 0, 0, 0, 0, 0};
    __builtin_amdgcn_tensor_load_to_lds(g0, g1, gz, gz, gz8, 0);
#else
    __builtin_amdgcn_tensor_load_to_lds(g0, g1, gz, gz, 0);
#endif
}
#endif

// ---------------------------------------------------------------------------
// WMMA GEMM:  C[m,n] = sum_k A[m,k] * Wt[n,k]  (+bias[n]) (+res[m,n])
// A: bf16 row-major (lda), Wt: bf16 row-major N x K (ldb), C: fp32 (ldc).
// Block = 256 threads = 8 waves; block tile 128(M) x 64(N); each wave 16x64.
// Weight tile for the block is staged into LDS (TDM DMA, padded pitch) per
// K-chunk of KC; all 8 waves read fragments with ds_load_b128.  All four
// B fragments are fetched before the 4-WMMA chain so the scheduler can use
// partial s_wait_dscnt and overlap LDS reads with the matrix pipe.
// Grid: x = N/64, y = M/128 (exact).  K, KC multiples of 32, K % KC == 0.
// ---------------------------------------------------------------------------
union FragBF { v16bf v; uint4 q[2]; };

__global__ __launch_bounds__(256)
void gemm_bf16_wmma(const unsigned short* __restrict__ A, int lda,
                    const unsigned short* __restrict__ Wt, int ldb,
                    const float* __restrict__ bias,
                    const float* __restrict__ res, int ldr,
                    float* __restrict__ Cout, int ldc,
                    int M, int Nn, int K, int KC)
{
    __shared__ unsigned short smemB[64 * (256 + 8)];   // max KC=256, pitch KC+8

    const int lane = threadIdx.x & 31;
    const int wave = threadIdx.x >> 5;
    const int half = lane >> 4;          // 0: lanes 0-15, 1: lanes 16-31
    const int lr   = lane & 15;
    const int m0   = (blockIdx.y * 8 + wave) * 16;
    const int n0   = blockIdx.x * 64;
    const int pitch = KC + 8;            // elements per LDS row (16B pad)

    v8f acc0 = {}, acc1 = {}, acc2 = {}, acc3 = {};
    const size_t arow = (size_t)(m0 + lr) * lda;

    for (int kc = 0; kc < K; kc += KC) {
        // ---- stage this block's 64 x KC weight tile into LDS ----
#ifdef USE_TDM
        if (wave == 0) {
            // single __shared__ allocation -> LDS offset 0
            tdm_load_tile(Wt + (size_t)n0 * ldb + kc, 0u, 64, KC, ldb);
            __builtin_amdgcn_s_wait_tensorcnt(0);
        }
        // The TDM writes LDS behind the compiler's back: without a visible
        // store, LLVM folds smemB loads to undef and deletes the ds_loads.
        // (a) opaque never-taken store keeps the array "written";
        // (b) memory clobber pins load ordering below the staging point.
        if (__builtin_expect(blockDim.y == 2, 0)) smemB[threadIdx.x] = 0;
        asm volatile("" ::: "memory");
#else
        {   // cooperative fallback: 16B per thread per step, padded pitch
            const int vecs_per_row = KC >> 3;            // 8 elems per 16B
            for (int idx = threadIdx.x; idx < 64 * vecs_per_row; idx += 256) {
                int row = idx / vecs_per_row;
                int kk  = (idx % vecs_per_row) << 3;
                *reinterpret_cast<uint4*>(smemB + row * pitch + kk) =
                    *reinterpret_cast<const uint4*>(Wt + (size_t)(n0 + row) * ldb + kc + kk);
            }
        }
#endif
        __syncthreads();

        if (kc + KC < K)   // prefetch next A chunk (global_prefetch_b8)
            __builtin_prefetch(A + arow + kc + KC, 0, 0);

        for (int kb = 0; kb < KC; kb += 32) {
            // A fragment (16-bit A 16x32, contiguous k order shared with B):
            //  lanes 0-15 -> K[+0..15]; lanes 16-31 -> K[+16..31]
            FragBF a;
            const int ka = kc + kb + (half ? 16 : 0);
            a.q[0] = *reinterpret_cast<const uint4*>(A + arow + ka);
            a.q[1] = *reinterpret_cast<const uint4*>(A + arow + ka + 8);

            // All four B fragments from LDS first (8x ds_load_b128), so the
            // scheduler can overlap LDS fetch with the WMMA chain.
            // B layout: lanes 0-15 hold K[+0..15], lanes 16-31 K[+16..31].
            const int kbb = kb + (half ? 16 : 0);
            FragBF bf0, bf1, bf2, bf3;
            {
                const unsigned short* s0 = smemB + (0 * 16 + lr) * pitch + kbb;
                const unsigned short* s1 = smemB + (1 * 16 + lr) * pitch + kbb;
                const unsigned short* s2 = smemB + (2 * 16 + lr) * pitch + kbb;
                const unsigned short* s3 = smemB + (3 * 16 + lr) * pitch + kbb;
                bf0.q[0] = *reinterpret_cast<const uint4*>(s0);
                bf0.q[1] = *reinterpret_cast<const uint4*>(s0 + 8);
                bf1.q[0] = *reinterpret_cast<const uint4*>(s1);
                bf1.q[1] = *reinterpret_cast<const uint4*>(s1 + 8);
                bf2.q[0] = *reinterpret_cast<const uint4*>(s2);
                bf2.q[1] = *reinterpret_cast<const uint4*>(s2 + 8);
                bf3.q[0] = *reinterpret_cast<const uint4*>(s3);
                bf3.q[1] = *reinterpret_cast<const uint4*>(s3 + 8);
            }
            acc0 = __builtin_amdgcn_wmma_f32_16x16x32_bf16(
                       false, a.v, false, bf0.v, (short)0, acc0, false, false);
            acc1 = __builtin_amdgcn_wmma_f32_16x16x32_bf16(
                       false, a.v, false, bf1.v, (short)0, acc1, false, false);
            acc2 = __builtin_amdgcn_wmma_f32_16x16x32_bf16(
                       false, a.v, false, bf2.v, (short)0, acc2, false, false);
            acc3 = __builtin_amdgcn_wmma_f32_16x16x32_bf16(
                       false, a.v, false, bf3.v, (short)0, acc3, false, false);
        }
        __syncthreads();
    }

#pragma unroll
    for (int t = 0; t < 4; ++t) {
        const v8f& acc = (t == 0) ? acc0 : (t == 1) ? acc1 : (t == 2) ? acc2 : acc3;
        const int n = n0 + t * 16 + lr;
        const float bv = bias ? bias[n] : 0.f;
#pragma unroll
        for (int r = 0; r < 8; ++r) {
            const int m = m0 + half * 8 + r;
            float v = acc[r] + bv;
            if (res) v += res[(size_t)m * ldr + n];
            Cout[(size_t)m * ldc + n] = v;
        }
    }
}

// ---------------------------------------------------------------------------
// Fused layernorm over C=256 per token, with optional elementwise pre-scale:
//   v = a[token,c] * (mul?mul[token,c]:1) * (skip?skip[0]:1) * (gate?gate[b,c]:1)
//   out = LN(v)*w + b   -> optional fp32 out + optional bf16 out
// ---------------------------------------------------------------------------
__global__ __launch_bounds__(256)
void fused_ln256_kernel(const float* __restrict__ a,
                        const float* __restrict__ mul,
                        const float* __restrict__ gate,
                        const float* __restrict__ skip,
                        const float* __restrict__ w,
                        const float* __restrict__ bvec,
                        float* __restrict__ of32,
                        unsigned short* __restrict__ obf)
{
    const int token = blockIdx.x;
    const int c = threadIdx.x;
    const size_t idx = (size_t)token * kC + c;
    float v = a[idx];
    if (mul)  v *= mul[idx];
    if (skip) v *= skip[0];
    if (gate) { int b = token / kN; v *= gate[b * kC + c]; }

    __shared__ float s1[256], s2[256];
    s1[c] = v; s2[c] = v * v;
    __syncthreads();
    for (int o = 128; o > 0; o >>= 1) {
        if (c < o) { s1[c] += s1[c + o]; s2[c] += s2[c + o]; }
        __syncthreads();
    }
    const float m   = s1[0] * (1.f / kC);
    const float var = s2[0] * (1.f / kC) - m * m;
    const float inv = rsqrtf(var + 1e-5f);
    const float o   = (v - m) * inv * w[c] + bvec[c];
    if (of32) of32[idx] = o;
    if (obf)  obf[idx]  = f2bf(o);
}

// ---------------------------------------------------------------------------
// z_avg[b,c] = mean_n xn[b,n,c]
// ---------------------------------------------------------------------------
__global__ void zavg_kernel(const float* __restrict__ xn, float* __restrict__ zavg)
{
    int idx = blockIdx.x * 256 + threadIdx.x;      // B*C = 4096
    if (idx >= kB * kC) return;
    int b = idx >> 8, c = idx & 255;
    const float* p = xn + (size_t)b * kN * kC + c;
    float s = 0.f;
    for (int n = 0; n < kN; ++n) s += p[(size_t)n * kC];
    zavg[idx] = s * (1.f / kN);
}

// ---------------------------------------------------------------------------
// gate[b,c] = sigmoid( zavg[b]·fc_w[c] + fc_b[c] + conv1d(zavg, ca_w)[c] )
// ---------------------------------------------------------------------------
__global__ __launch_bounds__(256)
void gate_kernel(const float* __restrict__ zavg, const float* __restrict__ fc_w,
                 const float* __restrict__ fc_b, const float* __restrict__ ca_w,
                 float* __restrict__ gate)
{
    const int b = blockIdx.x, c = threadIdx.x;
    __shared__ float zs[256];
    zs[c] = zavg[b * kC + c];
    __syncthreads();
    float acc = fc_b[c];
    for (int k = 0; k < kC; ++k) acc += zs[k] * fc_w[c * kC + k];
    for (int j = 0; j < 3; ++j) {
        int cc = c + j - 1;
        if (cc >= 0 && cc < kC) acc += zs[cc] * ca_w[j];
    }
    gate[b * kC + c] = sigmoidf_(acc);
}

// ---------------------------------------------------------------------------
// depthwise 3x3 conv + SiLU on xc (low 64 channels of xz), written into u in
// scan order: g<2 -> l=h*W+w ; g>=2 -> l=w*H+h   (u layout: [(b*Cg+c)*L + l])
// ---------------------------------------------------------------------------
__global__ void dwconv_silu_u_kernel(const float* __restrict__ xz,
                                     const float* __restrict__ cw,   // [Cg*9]
                                     const float* __restrict__ cb,   // [Cg]
                                     float* __restrict__ u, int g)
{
    long long idx = (long long)blockIdx.x * 256 + threadIdx.x;  // B*N*Cg
    if (idx >= kMT * kCg) return;
    int c = (int)(idx & 63);
    long long t = idx >> 6;
    int w = (int)(t % kW); t /= kW;
    int h = (int)(t % kH);
    int b = (int)(t / kH);
    float acc = cb[c];
#pragma unroll
    for (int ky = 0; ky < 3; ++ky) {
        int hh = h + ky - 1;
        if (hh < 0 || hh >= kH) continue;
#pragma unroll
        for (int kx = 0; kx < 3; ++kx) {
            int ww = w + kx - 1;
            if (ww < 0 || ww >= kW) continue;
            acc += xz[((size_t)b * kN + hh * kW + ww) * 128 + c] * cw[c * 9 + ky * 3 + kx];
        }
    }
    int l = (g < 2) ? (h * kW + w) : (w * kH + h);
    u[((size_t)(b * kCg + c)) * kL + l] = siluf_(acc);
}

// ---------------------------------------------------------------------------
// x_dbl[b,r,l] = sum_c u[b,c,l] * x_proj_w[r,c]    (r = 0..5)
// ---------------------------------------------------------------------------
__global__ void xdbl_kernel(const float* __restrict__ u,
                            const float* __restrict__ xpw,   // [6*Cg]
                            float* __restrict__ xdbl)
{
    int idx = blockIdx.x * 256 + threadIdx.x;   // B*L
    if (idx >= kB * kL) return;
    int b = idx / kL, l = idx % kL;
    float acc[6] = {0.f, 0.f, 0.f, 0.f, 0.f, 0.f};
    for (int c = 0; c < kCg; ++c) {
        float uv = u[((size_t)(b * kCg + c)) * kL + l];
#pragma unroll
        for (int r = 0; r < 6; ++r) acc[r] += uv * xpw[r * kCg + c];
    }
#pragma unroll
    for (int r = 0; r < 6; ++r) xdbl[((size_t)(b * 6 + r)) * kL + l] = acc[r];
}

// ---------------------------------------------------------------------------
// dt[b,c,l] = softplus( sum_r x_dbl[b,r,l]*dt_proj_w[c,r] + dt_proj_b[c] )
// ---------------------------------------------------------------------------
__global__ void dt_kernel(const float* __restrict__ xdbl,
                          const float* __restrict__ dtw,   // [Cg*4]
                          const float* __restrict__ dtb,   // [Cg]
                          float* __restrict__ dt)
{
    long long idx = (long long)blockIdx.x * 256 + threadIdx.x;   // B*Cg*L
    if (idx >= (long long)kB * kCg * kL) return;
    int l = (int)(idx % kL);
    long long t = idx / kL;
    int c = (int)(t % kCg);
    int b = (int)(t / kCg);
    float acc = dtb[c];
#pragma unroll
    for (int r = 0; r < 4; ++r)
        acc += xdbl[((size_t)(b * 6 + r)) * kL + l] * dtw[c * 4 + r];
    float sp = (acc > 20.f) ? acc : log1pf(expf(acc));
    dt[((size_t)(b * kCg + c)) * kL + l] = sp;
}

// ---------------------------------------------------------------------------
// Sequential selective scan (Ns = 1). Odd g scans in reverse (flip+flip-back).
//   h = exp(dt*A)*h + dt*Bs*u ; y = h*Cs + Dp*u
// ---------------------------------------------------------------------------
__global__ void scan_kernel(const float* __restrict__ dt,
                            const float* __restrict__ xdbl,
                            const float* __restrict__ u,
                            const float* __restrict__ A_log,  // [4*Cg]
                            const float* __restrict__ Dp,     // [4*Cg]
                            float* __restrict__ y, int g)
{
    int tid = blockIdx.x * blockDim.x + threadIdx.x;   // B*Cg = 1024
    if (tid >= kB * kCg) return;
    int b = tid >> 6, c = tid & 63;
    const float Ac = -expf(A_log[g * kCg + c]);
    const float Dc = Dp[g * kCg + c];
    const size_t base = (size_t)tid * kL;
    const size_t bB = (size_t)(b * 6 + 4) * kL;
    const size_t bC = (size_t)(b * 6 + 5) * kL;
    float h = 0.f;
    int l = (g & 1) ? (kL - 1) : 0;
    const int step = (g & 1) ? -1 : 1;
    for (int i = 0; i < kL; ++i, l += step) {
        float dtv = dt[base + l];
        float uv  = u[base + l];
        float a   = expf(dtv * Ac);
        h = a * h + dtv * xdbl[bB + l] * uv;
        y[base + l] = h * xdbl[bC + l] + Dc * uv;
    }
}

// ---------------------------------------------------------------------------
// out-norm (LN over Cg) at each pixel * silu(z), emit bf16 for out_proj GEMM
// ---------------------------------------------------------------------------
__global__ __launch_bounds__(64)
void outnorm_gate_kernel(const float* __restrict__ y,
                         const float* __restrict__ xz,
                         const float* __restrict__ onw,  // [Cg]
                         const float* __restrict__ onb,  // [Cg]
                         unsigned short* __restrict__ ypre, int g)
{
    const int token = blockIdx.x;           // B*N
    const int c = threadIdx.x;
    const int b = token / kN, n = token % kN;
    int l;
    if (g < 2) l = n;
    else { int h = n / kW, w = n % kW; l = w * kH + h; }

    float v = y[((size_t)(b * kCg + c)) * kL + l];
    __shared__ float s1[64], s2[64];
    s1[c] = v; s2[c] = v * v;
    __syncthreads();
    for (int o = 32; o > 0; o >>= 1) {
        if (c < o) { s1[c] += s1[c + o]; s2[c] += s2[c + o]; }
        __syncthreads();
    }
    const float m   = s1[0] * (1.f / kCg);
    const float var = s2[0] * (1.f / kCg) - m * m;
    const float inv = rsqrtf(var + 1e-5f);
    const float zv  = xz[(size_t)token * 128 + 64 + c];
    const float o   = ((v - m) * inv * onw[c] + onb[c]) * siluf_(zv);
    ypre[(size_t)token * kCg + c] = f2bf(o);
}

// ---------------------------------------------------------------------------
// MLP depthwise 3x3 conv + exact GELU, emit bf16 for fc2 GEMM
// ---------------------------------------------------------------------------
__global__ void dwconv_gelu_kernel(const float* __restrict__ hb,
                                   const float* __restrict__ dww,  // [HID*9]
                                   const float* __restrict__ dwb,  // [HID]
                                   unsigned short* __restrict__ h2)
{
    long long idx = (long long)blockIdx.x * 256 + threadIdx.x;   // B*N*HID
    if (idx >= kMT * kHID) return;
    int c = (int)(idx & (kHID - 1));
    long long t = idx >> 10;
    int n = (int)(t % kN);
    int b = (int)(t / kN);
    int h = n / kW, w = n % kW;
    float acc = dwb[c];
#pragma unroll
    for (int ky = 0; ky < 3; ++ky) {
        int hh = h + ky - 1;
        if (hh < 0 || hh >= kH) continue;
#pragma unroll
        for (int kx = 0; kx < 3; ++kx) {
            int ww = w + kx - 1;
            if (ww < 0 || ww >= kW) continue;
            acc += hb[((size_t)b * kN + hh * kW + ww) * kHID + c] * dww[c * 9 + ky * 3 + kx];
        }
    }
    float gl = 0.5f * acc * (1.f + erff(acc * 0.70710678118654752f));
    h2[idx] = f2bf(gl);
}

// ---------------------------------------------------------------------------
// host
// ---------------------------------------------------------------------------
extern "C" void kernel_launch(void* const* d_in, const int* in_sizes, int n_in,
                              void* d_out, int out_size, void* d_ws, size_t ws_size,
                              hipStream_t stream)
{
    (void)in_sizes; (void)n_in; (void)out_size; (void)ws_size;
    const float* x        = (const float*)d_in[0];
    const float* norm1_w  = (const float*)d_in[1];
    const float* norm1_b  = (const float*)d_in[2];
    const float* fc_w     = (const float*)d_in[3];
    const float* fc_b     = (const float*)d_in[4];
    const float* ca_w     = (const float*)d_in[5];
    const float* skip_s   = (const float*)d_in[6];
    const float* proj_w   = (const float*)d_in[7];
    const float* proj_b   = (const float*)d_in[8];
    const float* in_proj_w= (const float*)d_in[9];
    const float* conv_w   = (const float*)d_in[10];
    const float* conv_b   = (const float*)d_in[11];
    const float* x_proj_w = (const float*)d_in[12];
    const float* dt_proj_w= (const float*)d_in[13];
    const float* dt_proj_b= (const float*)d_in[14];
    const float* A_log    = (const float*)d_in[15];
    const float* Dp       = (const float*)d_in[16];
    const float* onw      = (const float*)d_in[17];
    const float* onb      = (const float*)d_in[18];
    const float* out_proj_w=(const float*)d_in[19];
    const float* norm2_w  = (const float*)d_in[20];
    const float* norm2_b  = (const float*)d_in[21];
    const float* fc1_w    = (const float*)d_in[22];
    const float* fc1_b    = (const float*)d_in[23];
    const float* dw_w     = (const float*)d_in[24];
    const float* dw_b     = (const float*)d_in[25];
    const float* fc2_w    = (const float*)d_in[26];
    const float* fc2_b    = (const float*)d_in[27];
    float* out            = (float*)d_out;

    // ---- workspace layout (compile-time sizes, 256B aligned) ----
    char* base = (char*)d_ws;
    size_t off = 0;
    auto alloc = [&](size_t bytes) -> char* {
        char* p = base + off;
        off += (bytes + 255) & ~(size_t)255;
        return p;
    };
    float*          xn      = (float*)alloc(kMT * kC * 4);
    unsigned short* actbf   = (unsigned short*)alloc(kMT * kC * 2);   // reused 3x
    float*          zavg    = (float*)alloc(kB * kC * 4);
    float*          gate    = (float*)alloc(kB * kC * 4);
    float*          ym      = (float*)alloc(kMT * kC * 4);
    float*          x1      = (float*)alloc(kMT * kC * 4);
    unsigned short* inpj_bf = (unsigned short*)alloc(4 * 128 * 64 * 2);
    unsigned short* outp_bf = (unsigned short*)alloc(4 * 64 * 64 * 2);
    unsigned short* proj_bf = (unsigned short*)alloc(256 * 256 * 2);
    unsigned short* fc1_bf  = (unsigned short*)alloc(1024 * 256 * 2);
    unsigned short* fc2_bf  = (unsigned short*)alloc(256 * 1024 * 2);

    // scan-phase buffers, later overlaid by the MLP activations
    size_t phase = off;
    float*          xz   = (float*)alloc(kMT * 128 * 4);
    float*          ubuf = (float*)alloc((size_t)kB * kCg * kL * 4);
    float*          xdbl = (float*)alloc((size_t)kB * 6 * kL * 4);
    float*          dtb  = (float*)alloc((size_t)kB * kCg * kL * 4);
    float*          ybuf = (float*)alloc((size_t)kB * kCg * kL * 4);
    unsigned short* ypre = (unsigned short*)alloc(kMT * kCg * 2);
    // MLP overlay (used strictly after scan-phase buffers are dead)
    float*          hbuf = (float*)(base + phase);
    unsigned short* h2   = (unsigned short*)(base + phase + (size_t)kMT * kHID * 4 + 256);

    const int M = (int)kMT;
    auto gemm = [&](const unsigned short* A, int lda,
                    const unsigned short* Wt, int ldb,
                    const float* bias, const float* res, int ldr,
                    float* C, int ldc, int Nout, int K) {
        const int KC = (K >= 256) ? 256 : K;
        dim3 grid(Nout / 64, M / 128);
        gemm_bf16_wmma<<<grid, 256, 0, stream>>>(A, lda, Wt, ldb, bias, res, ldr,
                                                 C, ldc, M, Nout, K, KC);
    };

    // --- weight conversions ---
    cvt_bf16_kernel<<<(4*128*64 + 255)/256, 256, 0, stream>>>(in_proj_w, inpj_bf, 4*128*64);
    cvt_bf16_kernel<<<(4*64*64  + 255)/256, 256, 0, stream>>>(out_proj_w, outp_bf, 4*64*64);
    cvt_bf16_kernel<<<(256*256  + 255)/256, 256, 0, stream>>>(proj_w, proj_bf, 256*256);
    cvt_bf16_kernel<<<(1024*256 + 255)/256, 256, 0, stream>>>(fc1_w, fc1_bf, 1024*256);
    cvt_bf16_kernel<<<(256*1024 + 255)/256, 256, 0, stream>>>(fc2_w, fc2_bf, 256*1024);

    // --- ln1: xn (f32 + bf16) ---
    fused_ln256_kernel<<<M, 256, 0, stream>>>(x, nullptr, nullptr, nullptr,
                                              norm1_w, norm1_b, xn, actbf);
    // --- channel gate ---
    zavg_kernel<<<(kB*kC + 255)/256, 256, 0, stream>>>(xn, zavg);
    gate_kernel<<<kB, 256, 0, stream>>>(zavg, fc_w, fc_b, ca_w, gate);

    // --- four scan directions ---
    for (int g = 0; g < 4; ++g) {
        // in_proj: (M x 64) @ (128 x 64)^T -> xz (M x 128)
        gemm(actbf + g * kCg, kC, inpj_bf + g * 128 * 64, 64,
             nullptr, nullptr, 0, xz, 128, 128, 64);
        dwconv_silu_u_kernel<<<(int)((kMT*kCg + 255)/256), 256, 0, stream>>>(
            xz, conv_w + g * kCg * 9, conv_b + g * kCg, ubuf, g);
        xdbl_kernel<<<(kB*kL + 255)/256, 256, 0, stream>>>(
            ubuf, x_proj_w + g * 6 * kCg, xdbl);
        dt_kernel<<<(int)(((long long)kB*kCg*kL + 255)/256), 256, 0, stream>>>(
            xdbl, dt_proj_w + g * kCg * 4, dt_proj_b + g * kCg, dtb);
        scan_kernel<<<(kB*kCg + 255)/256, 256, 0, stream>>>(
            dtb, xdbl, ubuf, A_log, Dp, ybuf, g);
        outnorm_gate_kernel<<<M, 64, 0, stream>>>(
            ybuf, xz, onw + g * kCg, onb + g * kCg, ypre, g);
        // out_proj: (M x 64) @ (64 x 64)^T -> ym columns [g*64, g*64+64)
        gemm(ypre, kCg, outp_bf + g * 64 * 64, 64,
             nullptr, nullptr, 0, ym + g * kCg, kC, 64, 64);
    }

    // --- combine: LN(ym * skip * xn * gate) -> bf16 ---
    fused_ln256_kernel<<<M, 256, 0, stream>>>(ym, xn, gate, skip_s,
                                              norm1_w, norm1_b, nullptr, actbf);
    // --- proj (+bias, +residual x) -> x1 ---
    gemm(actbf, kC, proj_bf, kC, proj_b, x, kC, x1, kC, kC, kC);

    // --- MLP ---
    fused_ln256_kernel<<<M, 256, 0, stream>>>(x1, nullptr, nullptr, nullptr,
                                              norm2_w, norm2_b, nullptr, actbf);
    gemm(actbf, kC, fc1_bf, kC, fc1_b, nullptr, 0, hbuf, kHID, kHID, kC);
    dwconv_gelu_kernel<<<(int)((kMT*kHID + 255)/256), 256, 0, stream>>>(
        hbuf, dw_w, dw_b, h2);
    gemm(h2, kHID, fc2_bf, kHID, fc2_b, x1, kC, out, kC, kC, kHID);
}
